// CausalAttention_84018150245153
// MI455X (gfx1250) — matrix operand order repaired and checked by reading, not compiled, not run on hardware
//
#include <hip/hip_runtime.h>
#include <hip/hip_bf16.h>

typedef __bf16 bf16;
typedef __attribute__((ext_vector_type(16))) __bf16 v16bf;
typedef __attribute__((ext_vector_type(8)))  float  v8f;

// Problem dims (fixed by the reference).
static constexpr int Bn = 4;
static constexpr int Sn = 2048;
static constexpr int Dn = 1024;
static constexpr float SCALE = 0.03125f; // 1/sqrt(1024)

// ---------------------------------------------------------------------------
// Workspace layout (bytes)
// ---------------------------------------------------------------------------
static constexpr size_t XB_OFF = 0;                                   // x bf16: 8192x1024
static constexpr size_t WQ_OFF = XB_OFF + (size_t)Bn*Sn*Dn*2;
static constexpr size_t WK_OFF = WQ_OFF + (size_t)Dn*Dn*2;
static constexpr size_t WV_OFF = WK_OFF + (size_t)Dn*Dn*2;
static constexpr size_t QB_OFF = WV_OFF + (size_t)Dn*Dn*2;            // Q bf16
static constexpr size_t KB_OFF = QB_OFF + (size_t)Bn*Sn*Dn*2;         // K bf16
static constexpr size_t VT_OFF = KB_OFF + (size_t)Bn*Sn*Dn*2;         // V^T bf16: [b][e][s]
static constexpr size_t SC_OFF = VT_OFF + (size_t)Bn*Sn*Dn*2;         // scores f32: B*S*S
static constexpr size_t P_OFF  = SC_OFF + (size_t)Bn*Sn*Sn*4;         // P bf16: B*S*S

// ---------------------------------------------------------------------------
// WMMA fragment loaders (CDNA5 §7.12.2 VGPR layouts, wave32)
//   A: 16xK tile of row-major K-contiguous matrix (ld elements)
//      lane(m=lane&15, g=lane>>4): elems 0-7 = K[g*8 .. +7], 8-15 = K[16+g*8 ..]
//   B: (K=32)x16 tile where B[k,n] = R[n,k], R row-major K-contiguous
//      lane(n=lane&15, g=lane>>4): elems 0-7 = K[g*8 .. +7], 8-15 = K[16+g*8 ..] of row n
// ---------------------------------------------------------------------------
__device__ __forceinline__ v16bf load_frag_a(const bf16* __restrict__ base, int ld) {
    int lane = threadIdx.x & 31;
    int m = lane & 15, g = lane >> 4;
    const bf16* p = base + (size_t)m * ld + g * 8;
    v16bf f;
    ((uint4*)&f)[0] = *(const uint4*)(p);
    ((uint4*)&f)[1] = *(const uint4*)(p + 16);
    return f;
}

__device__ __forceinline__ v16bf load_frag_b(const bf16* __restrict__ base, int ld) {
    int lane = threadIdx.x & 31;
    int n = lane & 15, g = lane >> 4;
    const bf16* p = base + (size_t)n * ld + g * 8;
    v16bf f;
    ((uint4*)&f)[0] = *(const uint4*)(p);
    ((uint4*)&f)[1] = *(const uint4*)(p + 16);
    return f;
}

__device__ __forceinline__ v8f wmma_bf16(v16bf a, v16bf b, v8f c) {
    return __builtin_amdgcn_wmma_f32_16x16x32_bf16(false, a, false, b, (short)0, c,
                                                   false, false);
}

// 32(m) x 64(n) wave tile helpers ------------------------------------------
__device__ __forceinline__ void load_step(const bf16* __restrict__ Arow,
                                          const bf16* __restrict__ Brow,
                                          int lda, int ldb, int k0,
                                          v16bf a[2], v16bf b[4]) {
    a[0] = load_frag_a(Arow + k0, lda);
    a[1] = load_frag_a(Arow + (size_t)16 * lda + k0, lda);
    #pragma unroll
    for (int j = 0; j < 4; ++j)
        b[j] = load_frag_b(Brow + (size_t)(16 * j) * ldb + k0, ldb);
}

__device__ __forceinline__ void mma_step(v8f acc[2][4], const v16bf a[2], const v16bf b[4]) {
    #pragma unroll
    for (int i = 0; i < 2; ++i)
        #pragma unroll
        for (int j = 0; j < 4; ++j)
            acc[i][j] = wmma_bf16(a[i], b[j], acc[i][j]);
}

// K-loop, software-pipelined with 2x-unrolled ping-pong fragment buffers:
// no register rotation copies; next step's 12 b128 loads fly under the
// current step's 8 WMMAs.  N = K/32 may be odd -> parity-peeled tail.
__device__ __forceinline__ void gemm_core(v8f acc[2][4],
                                          const bf16* __restrict__ Arow,
                                          const bf16* __restrict__ Brow,
                                          int lda, int ldb, int K) {
    #pragma unroll
    for (int i = 0; i < 2; ++i)
        #pragma unroll
        for (int j = 0; j < 4; ++j)
            #pragma unroll
            for (int r = 0; r < 8; ++r) acc[i][j][r] = 0.0f;

    const int N = K >> 5;              // number of 32-wide K steps (>= 1)
    v16bf a0[2], b0[4], a1[2], b1[4];
    load_step(Arow, Brow, lda, ldb, 0, a0, b0);
    int s = 1;
    for (; s + 1 < N; s += 2) {
        // invariant: a0/b0 hold step s-1
        load_step(Arow, Brow, lda, ldb, s * 32, a1, b1);
        __builtin_prefetch(Arow + s * 32 + 64, 0, 0);   // global_prefetch_b8
        __builtin_prefetch(Brow + s * 32 + 64, 0, 0);
        mma_step(acc, a0, b0);
        load_step(Arow, Brow, lda, ldb, (s + 1) * 32, a0, b0);
        mma_step(acc, a1, b1);
    }
    if (s == N - 1) {                   // one unloaded step remains
        load_step(Arow, Brow, lda, ldb, s * 32, a1, b1);
        mma_step(acc, a0, b0);
        mma_step(acc, a1, b1);
    } else {                            // s == N: only the buffered step left
        mma_step(acc, a0, b0);
    }
}

// ---------------------------------------------------------------------------
// fp32 -> bf16 convert, 8 elements/thread (b128 in/out)
// ---------------------------------------------------------------------------
__global__ void cvt_f32_bf16_v8(const float* __restrict__ in, bf16* __restrict__ out,
                                int n8) {
    int i = blockIdx.x * blockDim.x + threadIdx.x;
    if (i >= n8) return;
    const float4* p = (const float4*)in + (size_t)i * 2;
    float4 f0 = p[0], f1 = p[1];
    union { bf16 h[8]; uint4 u; } r;
    r.h[0] = (bf16)f0.x; r.h[1] = (bf16)f0.y; r.h[2] = (bf16)f0.z; r.h[3] = (bf16)f0.w;
    r.h[4] = (bf16)f1.x; r.h[5] = (bf16)f1.y; r.h[6] = (bf16)f1.z; r.h[7] = (bf16)f1.w;
    ((uint4*)out)[i] = r.u;
}

// ---------------------------------------------------------------------------
// Projection GEMM: Y[m,n] = sum_k A[m,k] * W[n,k]   (M=8192, N=1024, K=1024)
// WRITE_T==0: Y bf16 row-major (ld=Dn).  WRITE_T==1: transposed per batch:
// Vt[b][n][s], b=m/Sn, s=m%Sn (ld=Sn).
// Block = 8 waves; wave tile = 32(m) x 64(n); block tile = 32 x 512.
// ---------------------------------------------------------------------------
template <int WRITE_T>
__global__ __launch_bounds__(256) void proj_gemm(const bf16* __restrict__ A,
                                                 const bf16* __restrict__ W,
                                                 bf16* __restrict__ Y) {
    int wave = threadIdx.x >> 5;
    int m0 = blockIdx.x * 32;
    int n0 = blockIdx.y * 512 + wave * 64;

    v8f acc[2][4];
    gemm_core(acc, A + (size_t)m0 * Dn, W + (size_t)n0 * Dn, Dn, Dn, Dn);

    int lane = threadIdx.x & 31;
    int nl = lane & 15, g = lane >> 4;
    #pragma unroll
    for (int i = 0; i < 2; ++i)
        #pragma unroll
        for (int j = 0; j < 4; ++j)
            #pragma unroll
            for (int r = 0; r < 8; ++r) {
                int m = m0 + 16 * i + r + 8 * g;
                int n = n0 + 16 * j + nl;
                bf16 v = (bf16)acc[i][j][r];
                if (WRITE_T) {
                    int b = m >> 11, s = m & (Sn - 1);
                    Y[((size_t)b * Dn + n) * Sn + s] = v;
                } else {
                    Y[(size_t)m * Dn + n] = v;
                }
            }
}

// ---------------------------------------------------------------------------
// Scores GEMM: Sc[b][q][k] = scale * sum_e Q[b,q,e] K[b,k,e], -3e38 for k>q.
// Wave tile = 32(q) x 64(k); wave tiles fully above the diagonal are skipped
// (uniform branch, EXEC stays all-ones around WMMA).
// ---------------------------------------------------------------------------
__global__ __launch_bounds__(256) void scores_gemm(const bf16* __restrict__ Q,
                                                   const bf16* __restrict__ Km,
                                                   float* __restrict__ Sc) {
    int b = blockIdx.z;
    int wave = threadIdx.x >> 5;
    int q0 = blockIdx.x * 32;
    int k0t = blockIdx.y * 512 + wave * 64;
    if (k0t > q0 + 31) return; // fully masked wave tile: never written, never read

    const bf16* Qb = Q + (size_t)b * Sn * Dn;
    const bf16* Kb = Km + (size_t)b * Sn * Dn;

    v8f acc[2][4];
    gemm_core(acc, Qb + (size_t)q0 * Dn, Kb + (size_t)k0t * Dn, Dn, Dn, Dn);

    float* ScB = Sc + (size_t)b * Sn * Sn;
    int lane = threadIdx.x & 31;
    int nl = lane & 15, g = lane >> 4;
    #pragma unroll
    for (int i = 0; i < 2; ++i)
        #pragma unroll
        for (int j = 0; j < 4; ++j)
            #pragma unroll
            for (int r = 0; r < 8; ++r) {
                int q = q0 + 16 * i + r + 8 * g;
                int k = k0t + 16 * j + nl;
                float v = acc[i][j][r] * SCALE;
                if (k > q) v = -3.0e38f; // exp(v - max) underflows to exactly 0
                ScB[(size_t)q * Sn + k] = v;
            }
}

// ---------------------------------------------------------------------------
// Row softmax: one block (256 thr) per row. Reads exactly the causal prefix
// the scores kernel wrote (64-wide tiles with k0 <= (q&~31)+31), exp values
// in registers, writes bf16 P, zero-fills the tail so P*V is dense.
// ---------------------------------------------------------------------------
__global__ __launch_bounds__(256) void row_softmax(const float* __restrict__ Sc,
                                                   bf16* __restrict__ P) {
    int row = blockIdx.x;           // 0 .. B*Sn-1
    int b = row >> 11;
    int q = row & (Sn - 1);
    const float* src = Sc + ((size_t)b * Sn + q) * Sn;
    bf16* dst = P + ((size_t)b * Sn + q) * Sn;

    int qt = q & ~31;
    int Kread = (qt + 32 + 63) & ~63;   // end of written 64-wide tile range
    if (Kread > Sn) Kread = Sn;

    int t = threadIdx.x;
    int lane = t & 31, wid = t >> 5;
    __shared__ float sred[8];

    float vals[8];
    int cnt = 0;
    float vmax = -3.4e38f;
    for (int k = t; k < Kread; k += 256) {
        float v = src[k];
        vals[cnt++] = v;
        vmax = fmaxf(vmax, v);
    }
    #pragma unroll
    for (int off = 16; off > 0; off >>= 1) vmax = fmaxf(vmax, __shfl_xor(vmax, off, 32));
    if (lane == 0) sred[wid] = vmax;
    __syncthreads();
    float rowmax = -3.4e38f;
    #pragma unroll
    for (int i = 0; i < 8; ++i) rowmax = fmaxf(rowmax, sred[i]);
    __syncthreads();

    float vsum = 0.0f;
    for (int i = 0; i < cnt; ++i) {
        vals[i] = __expf(vals[i] - rowmax);
        vsum += vals[i];
    }
    #pragma unroll
    for (int off = 16; off > 0; off >>= 1) vsum += __shfl_xor(vsum, off, 32);
    if (lane == 0) sred[wid] = vsum;
    __syncthreads();
    float rowsum = 0.0f;
    #pragma unroll
    for (int i = 0; i < 8; ++i) rowsum += sred[i];
    float inv = 1.0f / rowsum;

    int idx = 0;
    for (int k = t; k < Kread; k += 256) dst[k] = (bf16)(vals[idx++] * inv);
    for (int k = Kread + t; k < Sn; k += 256) dst[k] = (bf16)0.0f;
}

// ---------------------------------------------------------------------------
// Output GEMM: O[b][q][e] = sum_k P[b,q,k] * Vt[b,e,k]; K truncated causally
// (P is exactly zero for k >= qt+32 by construction).
// ---------------------------------------------------------------------------
__global__ __launch_bounds__(256) void pv_gemm(const bf16* __restrict__ P,
                                               const bf16* __restrict__ Vt,
                                               float* __restrict__ O) {
    int b = blockIdx.z;
    int wave = threadIdx.x >> 5;
    int q0 = blockIdx.x * 32;
    int n0 = blockIdx.y * 512 + wave * 64;

    const bf16* Pb = P + (size_t)b * Sn * Sn;
    const bf16* Vb = Vt + (size_t)b * Dn * Sn;

    int Kmax = q0 + 32;            // all later P entries are exactly zero
    if (Kmax > Sn) Kmax = Sn;

    v8f acc[2][4];
    gemm_core(acc, Pb + (size_t)q0 * Sn, Vb + (size_t)n0 * Sn, Sn, Sn, Kmax);

    int lane = threadIdx.x & 31;
    int nl = lane & 15, g = lane >> 4;
    #pragma unroll
    for (int i = 0; i < 2; ++i)
        #pragma unroll
        for (int j = 0; j < 4; ++j)
            #pragma unroll
            for (int r = 0; r < 8; ++r) {
                int q = q0 + 16 * i + r + 8 * g;
                int e = n0 + 16 * j + nl;
                O[((size_t)b * Sn + q) * Dn + e] = acc[i][j][r];
            }
}

// ---------------------------------------------------------------------------
// Launch
// ---------------------------------------------------------------------------
extern "C" void kernel_launch(void* const* d_in, const int* in_sizes, int n_in,
                              void* d_out, int out_size, void* d_ws, size_t ws_size,
                              hipStream_t stream) {
    const float* x  = (const float*)d_in[0];
    const float* Wq = (const float*)d_in[1];
    const float* Wk = (const float*)d_in[2];
    const float* Wv = (const float*)d_in[3];
    float* out = (float*)d_out;

    char* ws = (char*)d_ws;
    bf16* xb  = (bf16*)(ws + XB_OFF);
    bf16* wqb = (bf16*)(ws + WQ_OFF);
    bf16* wkb = (bf16*)(ws + WK_OFF);
    bf16* wvb = (bf16*)(ws + WV_OFF);
    bf16* qb  = (bf16*)(ws + QB_OFF);
    bf16* kb  = (bf16*)(ws + KB_OFF);
    bf16* vt  = (bf16*)(ws + VT_OFF);
    float* sc = (float*)(ws + SC_OFF);
    bf16* p   = (bf16*)(ws + P_OFF);

    // 1) convert inputs to bf16 (8 elems/thread)
    {
        int nx8 = (Bn * Sn * Dn) / 8;               // 1048576
        int nw8 = (Dn * Dn) / 8;                    // 131072
        cvt_f32_bf16_v8<<<(nx8 + 255) / 256, 256, 0, stream>>>(x,  xb,  nx8);
        cvt_f32_bf16_v8<<<(nw8 + 255) / 256, 256, 0, stream>>>(Wq, wqb, nw8);
        cvt_f32_bf16_v8<<<(nw8 + 255) / 256, 256, 0, stream>>>(Wk, wkb, nw8);
        cvt_f32_bf16_v8<<<(nw8 + 255) / 256, 256, 0, stream>>>(Wv, wvb, nw8);
    }

    // 2) projections: Q, K (row-major), V transposed per batch
    {
        dim3 grid((Bn * Sn) / 32, Dn / 512, 1);     // 256 x 2
        proj_gemm<0><<<grid, 256, 0, stream>>>(xb, wqb, qb);
        proj_gemm<0><<<grid, 256, 0, stream>>>(xb, wkb, kb);
        proj_gemm<1><<<grid, 256, 0, stream>>>(xb, wvb, vt);
    }

    // 3) masked, scaled scores
    {
        dim3 grid(Sn / 32, Sn / 512, Bn);           // 64 x 4 x 4
        scores_gemm<<<grid, 256, 0, stream>>>(qb, kb, sc);
    }

    // 4) row softmax -> bf16 P (dense: zeros in masked region)
    row_softmax<<<Bn * Sn, 256, 0, stream>>>(sc, p);

    // 5) O = P * V
    {
        dim3 grid(Sn / 32, Dn / 512, Bn);           // 64 x 2 x 4
        pv_gemm<<<grid, 256, 0, stream>>>(p, vt, out);
    }
}
